// NeighborhoodAttention2Head_6923487282335
// MI455X (gfx1250) — matrix operand order repaired and checked by reading, not compiled, not run on hardware
//
#include <hip/hip_runtime.h>

#define EMBED 256
#define TS 8          // samples per block
#define KNB 16
#define SCALE 0.125f

// LDS strides (elements): 16B-aligned rows, bank-spread
#define NE_STRIDE 264   // u16 (256 data + pad), 528 B/row
#define GI_STRIDE 536   // u16 (512 data + pad), 1072 B/row
#define MU_STRIDE 520   // u16 (512 data + pad), 1040 B/row
#define QB_STRIDE 136   // u16 (128 data + pad), 272 B/row
#define PB_STRIDE 264   // u16 (256 data + pad)
#define CT_STRIDE 272   // f32

// bf16 weight workspace offsets (elements)
#define WQ_OFF  0         // [128][256]
#define WKT_OFF 65536     // [2][256][64]  (W_k transposed: [h][c][a])
#define WP_OFF  98304     // [256][512]
#define WG_OFF  229376    // [256][512]
#define W_TOTAL 360448

typedef __attribute__((ext_vector_type(16))) __bf16 v16bf;
typedef __attribute__((ext_vector_type(2)))  __bf16 v2bf;
typedef __attribute__((ext_vector_type(8)))  float  v8f;

struct FragBF { union { v16bf v; unsigned u[8]; uint4 q[2]; }; };

__device__ inline unsigned short f2bf_rn(float f) {   // RTNE (prep kernel only)
  unsigned u = __builtin_bit_cast(unsigned, f);
  u += 0x7FFFu + ((u >> 16) & 1u);
  return (unsigned short)(u >> 16);
}
__device__ inline unsigned short f2bf_tr(float f) {   // truncate, 1 op
  return (unsigned short)(__builtin_bit_cast(unsigned, f) >> 16);
}

#if defined(__has_builtin)
#if __has_builtin(__builtin_amdgcn_cvt_pk_bf16_f32)
#define HAVE_CVT_PK_BF16 1
#endif
#if __has_builtin(__builtin_amdgcn_perm)
#define HAVE_PERM 1
#endif
#endif

// pack two f32 -> dword of 2 bf16 (lo = a, hi = b)
__device__ inline unsigned pk2bf(float a, float b) {
#if defined(HAVE_CVT_PK_BF16)
  auto r = __builtin_amdgcn_cvt_pk_bf16_f32(a, b);
  return __builtin_bit_cast(unsigned, r);
#elif defined(HAVE_PERM)
  // v_perm_b32: one instruction truncating pack (result bytes {b3,b2,a3,a2})
  return __builtin_amdgcn_perm(__builtin_bit_cast(unsigned, b),
                               __builtin_bit_cast(unsigned, a), 0x07060302u);
#else
  return (unsigned)f2bf_tr(a) | ((unsigned)f2bf_tr(b) << 16);
#endif
}

// dot of two bf16 pairs into f32 acc, written so clang can fuse fpext(bf16)
// operands into v_fma_mix_f32_bf16 (VOP3P op 61 on CDNA5)
__device__ inline float dot2bf(unsigned a, unsigned b, float acc) {
  v2bf va = __builtin_bit_cast(v2bf, a);
  v2bf vb = __builtin_bit_cast(v2bf, b);
  acc = fmaf((float)va.x, (float)vb.x, acc);
  acc = fmaf((float)va.y, (float)vb.y, acc);
  return acc;
}

__device__ inline v8f wmma_bf16(const FragBF& a, const FragBF& b, v8f c) {
  return __builtin_amdgcn_wmma_f32_16x16x32_bf16(false, a.v, false, b.v, (short)0, c, false, false);
}

// A fragment (16x32 bf16): lane m=l&15, half=l>>4; per lane 16B at ks*64+half*16 and +32
__device__ inline FragBF lds_loadA(const unsigned short* rowBase, int kByteOff) {
  FragBF f;
  const char* p = (const char*)rowBase + kByteOff;
  f.q[0] = *(const uint4*)p;
  f.q[1] = *(const uint4*)(p + 32);
  return f;
}
// B fragment (32x16 bf16): lane n=l&15, half=l>>4; row-major [n][k] weights
__device__ inline FragBF glb_loadB(const unsigned short* w, size_t elemOff) {
  FragBF f;
  const uint4* p = (const uint4*)(w + elemOff);
  f.q[0] = p[0];
  f.q[1] = p[2];
  return f;
}

__global__ void prep_weights_kernel(const float* __restrict__ wq, const float* __restrict__ wk,
                                    const float* __restrict__ wp, const float* __restrict__ wg,
                                    unsigned short* __restrict__ out) {
  int i = blockIdx.x * 256 + threadIdx.x;        // W_TOTAL = 1408 * 256 exact
  float v;
  if (i < WKT_OFF) {
    v = wq[i];
  } else if (i < WP_OFF) {                        // transpose W_k -> [h][c][a]
    int j = i - WKT_OFF;
    int h = j >> 14, r = j & 16383, c = r >> 6, a = r & 63;
    v = wk[(h << 14) + a * 256 + c];
  } else if (i < WG_OFF) {
    v = wp[i - WP_OFF];
  } else {
    v = wg[i - WG_OFF];
  }
  out[i] = f2bf_rn(v);
}

__global__ __launch_bounds__(256, 4)
void fused_attn_kernel(const float* __restrict__ center, const float* __restrict__ neigh,
                       const float* __restrict__ conf, const int* __restrict__ valid,
                       const unsigned short* __restrict__ wbf, float* __restrict__ out) {
  __shared__ __align__(16) unsigned short sNeigh[TS * KNB * NE_STRIDE]; // 67584 B
  __shared__ __align__(16) unsigned short sGate [TS * GI_STRIDE];       //  8576 B
  __shared__ __align__(16) unsigned short sQbf  [TS * QB_STRIDE];       //  2176 B
  __shared__ __align__(16) unsigned short sPbf  [TS * 2 * PB_STRIDE];   //  8448 B
  __shared__ __align__(16) float          sAttn [TS * 2 * KNB];         //  1024 B
  __shared__ __align__(16) float          sLogC [TS * KNB];             //   512 B
  __shared__ __align__(16) unsigned short sMulti[TS * MU_STRIDE];       //  8320 B
  __shared__ __align__(16) float          sCtx  [TS * CT_STRIDE];       //  8704 B
  // total = 105344 B -> 3 workgroups per WGP (memory/compute overlap)

  const unsigned short* wq  = wbf + WQ_OFF;
  const unsigned short* wkT = wbf + WKT_OFF;
  const unsigned short* wp  = wbf + WP_OFF;
  const unsigned short* wg  = wbf + WG_OFF;

  const int tid  = threadIdx.x;
  const int wave = tid >> 5;
  const int lane = tid & 31;
  const int half = lane >> 4;
  const int l16  = lane & 15;
  const int blk  = blockIdx.x;

  // ---------------- Stage inputs to LDS (f32 -> bf16, 1-op packs) ----------------
  {
    const float4* src = (const float4*)(neigh + (size_t)blk * (TS * KNB * EMBED));
    #pragma unroll 4
    for (int i = 0; i < 32; ++i) {                 // 8192 float4, coalesced
      int e4 = i * 256 + tid;
      float4 v4 = src[e4];
      int e = e4 << 2;
      int row = e >> 8, col = e & 255;
      unsigned* dst = (unsigned*)&sNeigh[row * NE_STRIDE + col];
      dst[0] = pk2bf(v4.x, v4.y);
      dst[1] = pk2bf(v4.z, v4.w);
    }
  }
  {
    const float4* src = (const float4*)(center + (size_t)blk * (TS * EMBED));
    const int* vm = valid + blk * TS;
    #pragma unroll
    for (int i = 0; i < 2; ++i) {
      int e4 = i * 256 + tid;
      float4 v4 = src[e4];
      int e = e4 << 2;
      int row = e >> 8, col = e & 255;
      float mk = vm[row] ? 1.f : 0.f;
      unsigned* dst = (unsigned*)&sGate[row * GI_STRIDE + col];
      dst[0] = pk2bf(v4.x * mk, v4.y * mk);
      dst[1] = pk2bf(v4.z * mk, v4.w * mk);
    }
  }
  if (tid < TS * KNB) {
    float c = conf[(size_t)blk * (TS * KNB) + tid];
    sLogC[tid] = __logf(fmaxf(c, 1e-8f));
  }
  __syncthreads();

  // ---------------- Phase Q: q = center x Wq^T  (M=16(8 real), N=128, K=256) ----------------
  {
    int n0 = wave * 16;
    v8f acc = {};
    const unsigned short* rowBase = &sGate[(l16 & 7) * GI_STRIDE];  // duplicate rows 8..15
    #pragma unroll
    for (int ks = 0; ks < 8; ++ks) {
      FragBF a = lds_loadA(rowBase, ks * 64 + half * 16);
      FragBF b = glb_loadB(wq, (size_t)(n0 + l16) * EMBED + ks * 32 + half * 8);
      acc = wmma_bf16(a, b, acc);
    }
    if (half == 0) {                                // rows 0..7 are real
      #pragma unroll
      for (int r = 0; r < 8; ++r)
        sQbf[r * QB_STRIDE + n0 + l16] = f2bf_tr(acc[r]);
    }
  }
  __syncthreads();

  // ---------------- Phase P2: p[s,h,:] = q[s,h,:] x WkT[h]  (per head: M=16(8), N=256, K=64) ----------------
  #pragma unroll
  for (int h = 0; h < 2; ++h) {
    #pragma unroll
    for (int t = 0; t < 2; ++t) {
      int c0 = (wave * 2 + t) * 16;
      v8f acc = {};
      const unsigned short* rowBase = &sQbf[(l16 & 7) * QB_STRIDE + h * 64];
      #pragma unroll
      for (int ks = 0; ks < 2; ++ks) {
        FragBF a = lds_loadA(rowBase, ks * 64 + half * 16);
        FragBF b = glb_loadB(wkT, (size_t)h * 16384 + (size_t)(c0 + l16) * 64 + ks * 32 + half * 8);
        acc = wmma_bf16(a, b, acc);
      }
      if (half == 0) {
        #pragma unroll
        for (int r = 0; r < 8; ++r)
          sPbf[(r * 2 + h) * PB_STRIDE + c0 + l16] = f2bf_tr(acc[r]);
      }
    }
  }
  __syncthreads();

  // ---------------- Phase S: scores = (p . neigh) * SCALE + log_conf ----------------
  {
    int s = tid >> 5, h = (tid >> 4) & 1, kn = tid & 15;
    const unsigned* pp = (const unsigned*)&sPbf[(s * 2 + h) * PB_STRIDE];
    const unsigned* nr = (const unsigned*)&sNeigh[(s * 16 + kn) * NE_STRIDE];
    float dot = 0.f;
    #pragma unroll 16
    for (int j = 0; j < 128; ++j)
      dot = dot2bf(pp[j], nr[j], dot);
    sAttn[(s * 2 + h) * 16 + kn] = dot * SCALE + sLogC[s * 16 + kn];
  }
  __syncthreads();

  // ---------------- softmax over K=16 (one thread per (s,h)) ----------------
  if (tid < TS * 2) {
    int s = tid >> 1, h = tid & 1;
    float* sc = &sAttn[(s * 2 + h) * 16];
    float mx = sc[0];
    #pragma unroll
    for (int k = 1; k < 16; ++k) mx = fmaxf(mx, sc[k]);
    float e[16]; float sum = 0.f;
    #pragma unroll
    for (int k = 0; k < 16; ++k) { e[k] = __expf(sc[k] - mx); sum += e[k]; }
    float inv = 1.f / sum;
    #pragma unroll
    for (int k = 0; k < 16; ++k) sc[k] = e[k] * inv;
  }
  __syncthreads();

  // ---------------- Phase C: ctx_h = attn x neigh -> multi (bf16) ----------------
  {
    int s = tid >> 5, d0 = (tid & 31) * 8;        // 32 threads/sample, 8 d each
    float acc0[8], acc1[8];
    #pragma unroll
    for (int i = 0; i < 8; ++i) { acc0[i] = 0.f; acc1[i] = 0.f; }
    for (int kn = 0; kn < KNB; ++kn) {
      float a0 = sAttn[(s * 2 + 0) * 16 + kn];
      float a1 = sAttn[(s * 2 + 1) * 16 + kn];
      uint4 w4 = *(const uint4*)&sNeigh[(s * 16 + kn) * NE_STRIDE + d0];
      unsigned wd[4] = {w4.x, w4.y, w4.z, w4.w};
      #pragma unroll
      for (int j = 0; j < 4; ++j) {
        v2bf w2 = __builtin_bit_cast(v2bf, wd[j]);
        acc0[2 * j]     = fmaf((float)w2.x, a0, acc0[2 * j]);
        acc0[2 * j + 1] = fmaf((float)w2.y, a0, acc0[2 * j + 1]);
        acc1[2 * j]     = fmaf((float)w2.x, a1, acc1[2 * j]);
        acc1[2 * j + 1] = fmaf((float)w2.y, a1, acc1[2 * j + 1]);
      }
    }
    unsigned* m0 = (unsigned*)&sMulti[s * MU_STRIDE + d0];
    unsigned* m1 = (unsigned*)&sMulti[s * MU_STRIDE + 256 + d0];
    #pragma unroll
    for (int j = 0; j < 4; ++j) {
      m0[j] = pk2bf(acc0[2 * j], acc0[2 * j + 1]);
      m1[j] = pk2bf(acc1[2 * j], acc1[2 * j + 1]);
    }
  }
  __syncthreads();

  // ---------------- Phase P: context = multi x Wproj^T  (M=16(8), N=256, K=512) ----------------
  #pragma unroll
  for (int t = 0; t < 2; ++t) {
    int n0 = (wave * 2 + t) * 16;
    v8f acc = {};
    const unsigned short* rowBase = &sMulti[(l16 & 7) * MU_STRIDE];
    for (int ks = 0; ks < 16; ++ks) {
      FragBF a = lds_loadA(rowBase, ks * 64 + half * 16);
      FragBF b = glb_loadB(wp, (size_t)(n0 + l16) * 512 + ks * 32 + half * 8);
      acc = wmma_bf16(a, b, acc);
    }
    if (half == 0) {
      #pragma unroll
      for (int r = 0; r < 8; ++r) {
        int n = n0 + l16;
        float c = acc[r];
        sGate[r * GI_STRIDE + 256 + n] = f2bf_tr(c);  // bf16 half of gate input
        sCtx[r * CT_STRIDE + n] = c;                  // f32 for final blend
      }
    }
  }
  __syncthreads();

  // ---------------- Phase G: gate GEMM + sigmoid + blend + store ----------------
  {
    const float* cbase = center + (size_t)blk * (TS * EMBED);
    const int* vm = valid + blk * TS;
    #pragma unroll
    for (int t = 0; t < 2; ++t) {
      int n0 = (wave * 2 + t) * 16;
      v8f acc = {};
      const unsigned short* rowBase = &sGate[(l16 & 7) * GI_STRIDE];
      for (int ks = 0; ks < 16; ++ks) {
        FragBF a = lds_loadA(rowBase, ks * 64 + half * 16);
        FragBF b = glb_loadB(wg, (size_t)(n0 + l16) * 512 + ks * 32 + half * 8);
        acc = wmma_bf16(a, b, acc);
      }
      if (half == 0) {
        #pragma unroll
        for (int r = 0; r < 8; ++r) {
          int n = n0 + l16;
          float g  = 1.f / (1.f + __expf(-acc[r]));
          float cf = cbase[r * EMBED + n] * (vm[r] ? 1.f : 0.f);
          float cx = sCtx[r * CT_STRIDE + n];
          out[(size_t)(blk * TS + r) * EMBED + n] = g * cf + (1.f - g) * cx;
        }
      }
    }
  }
}

extern "C" void kernel_launch(void* const* d_in, const int* in_sizes, int n_in,
                              void* d_out, int out_size, void* d_ws, size_t ws_size,
                              hipStream_t stream) {
  (void)in_sizes; (void)n_in; (void)out_size; (void)ws_size;
  const float* center = (const float*)d_in[0];
  const float* neigh  = (const float*)d_in[1];
  const float* conf   = (const float*)d_in[2];
  const int*   valid  = (const int*)d_in[3];
  const float* Wq = (const float*)d_in[4];
  const float* Wk = (const float*)d_in[5];
  const float* Wp = (const float*)d_in[6];
  const float* Wg = (const float*)d_in[7];

  unsigned short* wbf = (unsigned short*)d_ws;   // 360448 bf16 = 704 KB
  prep_weights_kernel<<<W_TOTAL / 256, 256, 0, stream>>>(Wq, Wk, Wp, Wg, wbf);
  fused_attn_kernel<<<65536 / TS, 256, 0, stream>>>(center, neigh, conf, valid,
                                                    wbf, (float*)d_out);
}
